// ROIHead_77446850282136
// MI455X (gfx1250) — compile-verified, matrix operand order checked by reading, not run on hardware
//
#include <hip/hip_runtime.h>

typedef unsigned short u16;
typedef __attribute__((ext_vector_type(16))) __bf16 v16bf;
typedef __attribute__((ext_vector_type(8)))  float  v8f;

#define R_TOT 256       // NUM * S
#define CCH   512
#define FDIM  37
#define PP    7
#define KDIM  25088     // CCH * PP * PP
#define NDIM  4096
#define NCLS_ 21
#define REG_N 84        // NCLS * 4
#define KC    256       // K chunk per staging round
#define NCHUNK (KDIM / KC)   // 98

// f32 -> bf16, round to nearest even
__device__ __forceinline__ u16 f2bf(float f) {
  unsigned u = __builtin_bit_cast(unsigned, f);
  u += 0x7fffu + ((u >> 16) & 1u);
  return (u16)(u >> 16);
}

// ---------------------------------------------------------------------------
// Kernel 1: RoIPool(max) -> pooled A matrix, stored row-major bf16 [256][25088]
// ---------------------------------------------------------------------------
__global__ void __launch_bounds__(256)
roi_pool_k(const float* __restrict__ x, const float* __restrict__ rois,
           const int* __restrict__ rois_idx, const int* __restrict__ im_size,
           u16* __restrict__ poolA) {
  const int r = blockIdx.x;
  const float sx = (float)FDIM / (float)im_size[1];
  const float sy = (float)FDIM / (float)im_size[0];
  const float x1 = roundf(rois[r * 4 + 0] * sx);
  const float y1 = roundf(rois[r * 4 + 1] * sy);
  const float x2 = roundf(rois[r * 4 + 2] * sx);
  const float y2 = roundf(rois[r * 4 + 3] * sy);
  const float rw = fmaxf(x2 - x1 + 1.f, 1.f);
  const float rh = fmaxf(y2 - y1 + 1.f, 1.f);
  const int img = rois_idx[r];
  const float* __restrict__ xb = x + (size_t)img * CCH * FDIM * FDIM;

  for (int j = threadIdx.x; j < KDIM; j += 256) {
    int c   = j / 49;
    int bin = j - c * 49;
    int ph  = bin / PP, pw = bin - ph * PP;
    float wsF = fminf(fmaxf(floorf((float)pw * rw / PP) + x1, 0.f), (float)FDIM);
    float weF = fminf(fmaxf(ceilf((float)(pw + 1) * rw / PP) + x1, 0.f), (float)FDIM);
    float hsF = fminf(fmaxf(floorf((float)ph * rh / PP) + y1, 0.f), (float)FDIM);
    float heF = fminf(fmaxf(ceilf((float)(ph + 1) * rh / PP) + y1, 0.f), (float)FDIM);
    int w0 = (int)wsF, w1 = (int)weF, h0 = (int)hsF, h1 = (int)heF;
    float m = -3.402823466e38f;
    const float* __restrict__ xc = xb + (size_t)c * FDIM * FDIM;
    for (int h = h0; h < h1; ++h)
      for (int w = w0; w < w1; ++w)
        m = fmaxf(m, xc[h * FDIM + w]);
    if (w0 >= w1 || h0 >= h1) m = 0.f;   // empty bin -> 0 (reference NEG->0)
    poolA[(size_t)r * KDIM + j] = f2bf(m);
  }
}

// ---------------------------------------------------------------------------
// Kernel 2: FC1 via bf16 WMMA, K chunked by 256 with ping-pong LDS staging.
// Block = 512 thr = 16 waves; block owns all M=256 rows x 16 columns.
// W_lin streamed from HBM exactly once (256 blocks x disjoint 16-col slabs).
// ---------------------------------------------------------------------------
__global__ void __launch_bounds__(512)
fc1_wmma_k(const u16* __restrict__ poolA, const float* __restrict__ W,
           const float* __restrict__ b,  float* __restrict__ feat) {
  __shared__ u16 ldsB[2][16 * KC];       // ping-pong, [n_local][k] bf16, 2x8KB

  const int n0   = blockIdx.x * 16;
  const int tid  = threadIdx.x;
  const int wave = tid >> 5;             // 0..15  (wave32)
  const int lane = tid & 31;
  const int m0   = wave * 16;
  const int hi   = lane >> 4;            // 0/1
  const int off  = hi * 8;               // K sub-offset per ISA 16-bit layout
  const int nloc = lane & 15;

  // staging mapping: thread t covers k = t>>1, columns n4..n4+7 (n4 = (t&1)*8)
  const int ks = tid >> 1;               // 0..255
  const int n4 = (tid & 1) * 8;
  const float* __restrict__ wsrc = W + (size_t)ks * NDIM + n0 + n4;

  const u16* __restrict__ arow =
      poolA + (size_t)(m0 + (lane & 15)) * KDIM + off;

  // ---- preload + stage chunk 0 into buffer 0 ----
  {
    float4 w0 = *reinterpret_cast<const float4*>(wsrc);
    float4 w1 = *reinterpret_cast<const float4*>(wsrc + 4);
    u16* d = &ldsB[0][n4 * KC + ks];
    d[0 * KC] = f2bf(w0.x); d[1 * KC] = f2bf(w0.y);
    d[2 * KC] = f2bf(w0.z); d[3 * KC] = f2bf(w0.w);
    d[4 * KC] = f2bf(w1.x); d[5 * KC] = f2bf(w1.y);
    d[6 * KC] = f2bf(w1.z); d[7 * KC] = f2bf(w1.w);
  }
  __syncthreads();

  v8f acc = {};
  for (int c = 0; c < NCHUNK; ++c) {
    const int k0 = c * KC;
    // issue next chunk's W loads early (clamped, branch-free tail)
    int knext = k0 + KC;
    if (knext > KDIM - KC) knext = KDIM - KC;
    const float* __restrict__ ws2 = wsrc + (size_t)knext * NDIM;
    float4 nw0 = *reinterpret_cast<const float4*>(ws2);
    float4 nw1 = *reinterpret_cast<const float4*>(ws2 + 4);
    // prefetch two chunks ahead (gfx1250 global_prefetch_b8)
    int kpf = k0 + 2 * KC;
    if (kpf > KDIM - KC) kpf = KDIM - KC;
    __builtin_prefetch(wsrc + (size_t)kpf * NDIM, 0, 1);

    // ---- 8 WMMAs on current buffer ----
    const u16* __restrict__ bb = &ldsB[c & 1][nloc * KC + off];
    const u16* __restrict__ aa = arow + k0;
#pragma unroll
    for (int s = 0; s < 8; ++s) {
      struct Q { uint4 a, b; } fa, fb;
      fa.a = *reinterpret_cast<const uint4*>(aa + s * 32);        // K off..off+7
      fa.b = *reinterpret_cast<const uint4*>(aa + s * 32 + 16);   // K 16+off..
      fb.a = *reinterpret_cast<const uint4*>(bb + s * 32);        // ds_load_b128
      fb.b = *reinterpret_cast<const uint4*>(bb + s * 32 + 16);
      v16bf av = __builtin_bit_cast(v16bf, fa);
      v16bf bv = __builtin_bit_cast(v16bf, fb);
      acc = __builtin_amdgcn_wmma_f32_16x16x32_bf16(false, av, false, bv,
                                                    (short)0, acc, false, false);
    }

    // ---- convert + store next chunk into the other buffer ----
    {
      u16* d = &ldsB[(c + 1) & 1][n4 * KC + ks];
      d[0 * KC] = f2bf(nw0.x); d[1 * KC] = f2bf(nw0.y);
      d[2 * KC] = f2bf(nw0.z); d[3 * KC] = f2bf(nw0.w);
      d[4 * KC] = f2bf(nw1.x); d[5 * KC] = f2bf(nw1.y);
      d[6 * KC] = f2bf(nw1.z); d[7 * KC] = f2bf(nw1.w);
    }
    __syncthreads();   // one barrier per 256-K chunk
  }

  const int   n    = n0 + nloc;
  const float bias = b[n];
#pragma unroll
  for (int rr = 0; rr < 8; ++rr) {       // C layout: VGPR rr -> M = rr (+8 hi)
    int   m = m0 + rr + hi * 8;
    float v = acc[rr] + bias;
    feat[(size_t)m * NDIM + n] = v > 0.f ? v : 0.f;
  }
}

// ---------------------------------------------------------------------------
// Kernel 3: reg/cls heads (tiny GEMMs). One block per ROI row; feat row in LDS.
// ---------------------------------------------------------------------------
__global__ void __launch_bounds__(128)
fc2_k(const float* __restrict__ feat,
      const float* __restrict__ Wreg, const float* __restrict__ breg,
      const float* __restrict__ Wcls, const float* __restrict__ bcls,
      float* __restrict__ out) {
  __shared__ float f[NDIM];
  const int m = blockIdx.x;
  for (int i = threadIdx.x; i < NDIM; i += 128)
    f[i] = feat[(size_t)m * NDIM + i];
  __syncthreads();
  for (int o = threadIdx.x; o < REG_N + NCLS_; o += 128) {
    if (o < REG_N) {
      float acc = breg[o];
      for (int k = 0; k < NDIM; ++k)
        acc = fmaf(f[k], Wreg[(size_t)k * REG_N + o], acc);
      out[(size_t)m * REG_N + o] = acc;
    } else {
      int o2 = o - REG_N;
      float acc = bcls[o2];
      for (int k = 0; k < NDIM; ++k)
        acc = fmaf(f[k], Wcls[(size_t)k * NCLS_ + o2], acc);
      out[(size_t)R_TOT * REG_N + (size_t)m * NCLS_ + o2] = acc;
    }
  }
}

// ---------------------------------------------------------------------------
extern "C" void kernel_launch(void* const* d_in, const int* in_sizes, int n_in,
                              void* d_out, int out_size, void* d_ws, size_t ws_size,
                              hipStream_t stream) {
  const float* x    = (const float*)d_in[0];
  const float* rois = (const float*)d_in[1];
  const int*   ridx = (const int*)d_in[2];
  const int*   imsz = (const int*)d_in[3];
  const float* Wlin = (const float*)d_in[4];
  const float* blin = (const float*)d_in[5];
  const float* Wreg = (const float*)d_in[6];
  const float* breg = (const float*)d_in[7];
  const float* Wcls = (const float*)d_in[8];
  const float* bcls = (const float*)d_in[9];
  float* out = (float*)d_out;

  // workspace: pooled A (bf16, 12.85 MB) then feat (f32, 4.19 MB)
  u16*   poolA = (u16*)d_ws;
  float* feat  = (float*)((char*)d_ws + (size_t)R_TOT * KDIM * sizeof(u16));

  roi_pool_k<<<R_TOT, 256, 0, stream>>>(x, rois, ridx, imsz, poolA);
  fc1_wmma_k<<<NDIM / 16, 512, 0, stream>>>(poolA, Wlin, blin, feat);
  fc2_k<<<R_TOT, 128, 0, stream>>>(feat, Wreg, breg, Wcls, bcls, out);
}
